// RNNDecoder_29497835389180
// MI455X (gfx1250) — compile-verified, matrix operand order checked by reading, not run on hardware
//
#include <hip/hip_runtime.h>

// ---------------------------------------------------------------------------
// MI455X (gfx1250, wave32) attention RNN decoder.
// - bf16 WMMA (v_wmma_f32_16x16x32_bf16) for all GEMMs, fp32 accumulate.
// - Weights converted once per call to bf16 [N][K] (pre-transposed) so WMMA
//   B fragments are two contiguous 16B loads per lane. Weight set (~59MB
//   bf16) is L2-resident (192MB L2) -> 128-step recurrence is L2-BW bound.
// - A panels (activations, shared by all 4 waves of a block) are staged into
//   LDS with the Tensor Data Mover (TENSOR_LOAD_TO_LDS + s_wait_tensorcnt),
//   removing 4x-redundant global A traffic; fragments then come from LDS
//   via ds_load_b128.
// ---------------------------------------------------------------------------

#define DEV __device__ __forceinline__

typedef __attribute__((ext_vector_type(16))) __bf16 v16bf;
typedef __attribute__((ext_vector_type(8)))  float  v8f;
typedef __attribute__((ext_vector_type(4)))  float  f32x4;
typedef __attribute__((ext_vector_type(4)))  unsigned int u32x4;
typedef __attribute__((ext_vector_type(4)))  int    i32x4;
typedef __attribute__((ext_vector_type(8)))  int    i32x8;

union FragBF { v16bf v; f32x4 q[2]; };

// Fragment loader for 16-bit 16x32 A tile / 32x16 B tile (wave32 layout):
// lane%16 = row (A: m, B: n), lane/16 selects K-half (0 or 8).
// VGPR 0-3 <- K[kh .. kh+7], VGPR 4-7 <- K[16+kh .. 16+kh+7].
DEV v16bf load_frag(const __bf16* base, int ld, int lane) {
  const int r  = lane & 15;
  const int kh = (lane & 16) >> 1;       // 0 or 8
  const __bf16* p = base + (size_t)r * ld + kh;
  FragBF f;
  f.q[0] = *reinterpret_cast<const f32x4*>(p);
  f.q[1] = *reinterpret_cast<const f32x4*>(p + 16);
  return f.v;
}

DEV v8f wmma_bf16(v16bf a, v16bf b, v8f c) {
  // (neg_a, A, neg_b, B, c_mod, C, reuse_a, reuse_b)
  return __builtin_amdgcn_wmma_f32_16x16x32_bf16(false, a, false, b,
                                                 (short)0, c, false, false);
}

// Dual-accumulator K loop: breaks WMMA->WMMA RAW chain for pipelining.
DEV void mma_loop(v8f& acc0, v8f& acc1, const __bf16* A, int lda,
                  const __bf16* BT, int ldb, int nkt, int lane) {
  int kt = 0;
  for (; kt + 1 < nkt; kt += 2) {
    acc0 = wmma_bf16(load_frag(A + kt * 32, lda, lane),
                     load_frag(BT + kt * 32, ldb, lane), acc0);
    acc1 = wmma_bf16(load_frag(A + (kt + 1) * 32, lda, lane),
                     load_frag(BT + (kt + 1) * 32, ldb, lane), acc1);
  }
  if (kt < nkt)
    acc0 = wmma_bf16(load_frag(A + kt * 32, lda, lane),
                     load_frag(BT + kt * 32, ldb, lane), acc0);
}

// ---- Tensor Data Mover: DMA a packed [rows x kcols] bf16 tile (row stride
// kcols elements in global) into LDS at lds_byte_addr. D# per ISA ch.8:
// group0 = {count=1 | type=2 | lds_addr | global_addr}, group1 = dims/strides.
DEV void tdm_load_tile(unsigned int lds_byte_addr, const __bf16* gptr,
                       int kcols, int rows) {
  unsigned long long ga = (unsigned long long)(uintptr_t)gptr;
  u32x4 g0;
  g0.x = 1u;                                           // count=1 (valid), user
  g0.y = lds_byte_addr;                                // lds_addr [63:32]
  g0.z = (unsigned int)ga;                             // global_addr lo
  g0.w = (unsigned int)((ga >> 32) & 0x01FFFFFFu) | (2u << 30); // addr hi|type=2
  i32x8 g1;
  g1[0] = (int)(1u << 16);                             // data_size=1 (2 bytes)
  g1[1] = (int)(((unsigned)kcols & 0xFFFFu) << 16);    // tensor_dim0 lo16
  g1[2] = (int)((((unsigned)kcols >> 16) & 0xFFFFu) |
                (((unsigned)rows & 0xFFFFu) << 16));   // dim0 hi | dim1 lo
  g1[3] = (int)((((unsigned)rows >> 16) & 0xFFFFu) |
                (((unsigned)kcols & 0xFFFFu) << 16));  // dim1 hi | tile_dim0
  g1[4] = (int)((unsigned)rows & 0xFFFFu);             // tile_dim1 (tile_dim2=0)
  g1[5] = kcols;                                       // tensor_dim0_stride lo32
  g1[6] = 0;
  g1[7] = 0;
  i32x4 z4 = {0, 0, 0, 0};
#if __clang_major__ >= 23
  i32x8 z8 = {0, 0, 0, 0, 0, 0, 0, 0};
  __builtin_amdgcn_tensor_load_to_lds(g0, g1, z4, z4, z8, 0);
#else
  __builtin_amdgcn_tensor_load_to_lds(g0, g1, z4, z4, 0);
#endif
}

DEV float sigmoidf_(float x) { return 1.0f / (1.0f + __expf(-x)); }

// ------------------------------- prep kernels ------------------------------

// out[n*K + k] = bf16(in[k*N + n])   (fp32 [K][N] -> bf16 [N][K])
__global__ void k_transpose_bf16(const float* __restrict__ in,
                                 __bf16* __restrict__ out,
                                 int K, int N, long total) {
  long i = (long)blockIdx.x * blockDim.x + threadIdx.x;
  if (i >= total) return;
  int n = (int)(i / K), k = (int)(i % K);
  out[i] = (__bf16)in[(size_t)k * N + n];
}

__global__ void k_cast_bf16(const float* __restrict__ in,
                            __bf16* __restrict__ out, long total) {
  long i = (long)blockIdx.x * blockDim.x + threadIdx.x;
  if (i >= total) return;
  out[i] = (__bf16)in[i];
}

// y_emb[t][b][e] = bf16(emb[ys[b][t]][e]);  time-major for per-step slices.
__global__ void k_gather_emb(const int* __restrict__ ys,
                             const float* __restrict__ emb,
                             __bf16* __restrict__ yemb) {
  int i = blockIdx.x * 256 + threadIdx.x;         // < 128*16*512
  int e = i & 511;
  int b = (i >> 9) & 15;
  int t = i >> 13;
  int tok = ys[b * 128 + t];
  yemb[i] = (__bf16)emb[(size_t)tok * 512 + e];
}

__global__ void k_init_state(float* c0, float* c1, float* aw,
                             __bf16* h0a, __bf16* h0b,
                             __bf16* h1a, __bf16* h1b, __bf16* ctxb) {
  int i = blockIdx.x * 256 + threadIdx.x;         // grid covers 16384
  if (i < 16 * 1024) {
    c0[i] = 0.f; c1[i] = 0.f;
    h0a[i] = (__bf16)0.f; h0b[i] = (__bf16)0.f;
    h1a[i] = (__bf16)0.f; h1b[i] = (__bf16)0.f;
  }
  if (i < 16 * 512) {
    aw[i] = 0.f; ctxb[i] = (__bf16)0.f;
  }
}

// enc_k[b*T+t][a] = enc_bf[b*T+t][:] @ WkT[a][:] + bk[a]    (M=8192,N=512,K=512)
__global__ __launch_bounds__(128) void k_enck(const __bf16* __restrict__ encB,
                                              const __bf16* __restrict__ WkT,
                                              const float* __restrict__ bk,
                                              float* __restrict__ encK) {
  int wave = threadIdx.x >> 5, lane = threadIdx.x & 31;
  int mt = blockIdx.x;                 // 0..511
  int nt = blockIdx.y * 4 + wave;      // 0..31
  const __bf16* Ab = encB + (size_t)mt * 16 * 512;
  const __bf16* Bb = WkT + (size_t)nt * 16 * 512;
  v8f a0 = {0,0,0,0,0,0,0,0}, a1 = {0,0,0,0,0,0,0,0};
  mma_loop(a0, a1, Ab, 512, Bb, 512, 16, lane);
  v8f acc = a0 + a1;
  int n = lane & 15, mo = (lane >> 4) * 8;
  float* out = encK + (size_t)mt * 16 * 512 + nt * 16;
  float bz = bk[nt * 16 + n];
#pragma unroll
  for (int v = 0; v < 8; ++v) out[(size_t)(mo + v) * 512 + n] = acc[v] + bz;
}

// ------------------------------ per-step kernels ---------------------------

// Fused LSTM layer: G = [A1|A2]@WxT' + hprev@WhT' + b, then cell update.
// grid.x = 64 (16-col groups of H=1024); 4 waves = gates i,f,g,o.
// A panel (16 x (K1+K2+H) bf16) staged to LDS once per block via TDM.
// Dynamic LDS: 16*(K1+K2+1024)*2 + 4KB gate-exchange.
__global__ __launch_bounds__(128) void k_lstm(const __bf16* __restrict__ A1, int K1,
                                              const __bf16* __restrict__ A2, int K2,
                                              const __bf16* __restrict__ hprev,
                                              const __bf16* __restrict__ WxT,
                                              const __bf16* __restrict__ WhT,
                                              const float* __restrict__ bias,
                                              float* __restrict__ cst,
                                              __bf16* __restrict__ hout) {
  const int H = 1024;
  extern __shared__ char smem[];
  __bf16* sA = (__bf16*)smem;                       // [16][K1+K2+H] packed
  float (*gsh)[16][16] =
      (float (*)[16][16])(smem + (size_t)16 * (K1 + K2 + H) * 2);

  int wave = threadIdx.x >> 5, lane = threadIdx.x & 31;
  int j0 = blockIdx.x * 16;
  int col = wave * H + j0;             // column in 4H gate matrix
  int ldx = K1 + K2;

  // One wave issues the TDM DMAs (EXEC ignored by TDM; wave-level op).
  unsigned int lbase = (unsigned int)(uintptr_t)sA;
  if (wave == 0) {
    tdm_load_tile(lbase, A1, K1, 16);
    if (A2) tdm_load_tile(lbase + (unsigned)(16 * K1 * 2), A2, K2, 16);
    tdm_load_tile(lbase + (unsigned)(16 * (K1 + K2) * 2), hprev, H, 16);
    __builtin_amdgcn_s_wait_tensorcnt(0);
  }
  __syncthreads();

  const __bf16* sA1 = sA;
  const __bf16* sA2 = sA + 16 * K1;
  const __bf16* sH  = sA + 16 * (K1 + K2);

  v8f a0 = {0,0,0,0,0,0,0,0}, a1 = {0,0,0,0,0,0,0,0};
  const __bf16* Bx = WxT + (size_t)col * ldx;
  mma_loop(a0, a1, sA1, K1, Bx, ldx, K1 >> 5, lane);
  if (K2) mma_loop(a0, a1, sA2, K2, Bx + K1, ldx, K2 >> 5, lane);
  const __bf16* Bh = WhT + (size_t)col * H;
  mma_loop(a0, a1, sH, H, Bh, H, H >> 5, lane);
  v8f acc = a0 + a1;

  int n = lane & 15, mo = (lane >> 4) * 8;
  float bz = bias[col + n];
#pragma unroll
  for (int v = 0; v < 8; ++v) gsh[wave][mo + v][n] = acc[v] + bz;
  __syncthreads();

  for (int idx = threadIdx.x; idx < 256; idx += 128) {
    int b = idx >> 4, j = idx & 15;
    float ig = sigmoidf_(gsh[0][b][j]);
    float fg = sigmoidf_(gsh[1][b][j]);
    float gg = tanhf(gsh[2][b][j]);
    float og = sigmoidf_(gsh[3][b][j]);
    int si = b * H + j0 + j;
    float c = fg * cst[si] + ig * gg;
    cst[si] = c;
    hout[si] = (__bf16)(og * tanhf(c));
  }
}

// Generic 16-row GEMM: D[16 x ncols] = act(A1@B1 + A2@B2 + bias).
// One WMMA tile column per wave; A panel staged to LDS via TDM.
// Dynamic LDS: 16*(K1+K2)*2 bytes.
__global__ __launch_bounds__(128) void k_gemm16(const __bf16* __restrict__ A1, int K1,
                                                const __bf16* __restrict__ A2, int K2,
                                                const __bf16* __restrict__ BT,
                                                const float* __restrict__ bias,
                                                float* __restrict__ outF, long ldF,
                                                __bf16* __restrict__ outB, int ldB16,
                                                int ncols, int act) {
  extern __shared__ char smemG[];
  __bf16* sA = (__bf16*)smemG;

  int wave = threadIdx.x >> 5, lane = threadIdx.x & 31;
  unsigned int lbase = (unsigned int)(uintptr_t)sA;
  if (wave == 0) {
    tdm_load_tile(lbase, A1, K1, 16);
    if (A2) tdm_load_tile(lbase + (unsigned)(16 * K1 * 2), A2, K2, 16);
    __builtin_amdgcn_s_wait_tensorcnt(0);
  }
  __syncthreads();

  int wv = blockIdx.x * 4 + wave;
  int col = wv * 16;
  if (col >= ncols) return;            // wave-uniform; after the barrier

  int ldb = K1 + K2;
  const __bf16* Bb = BT + (size_t)col * ldb;
  v8f a0 = {0,0,0,0,0,0,0,0}, a1 = {0,0,0,0,0,0,0,0};
  mma_loop(a0, a1, sA, K1, Bb, ldb, K1 >> 5, lane);
  if (K2) mma_loop(a0, a1, sA + 16 * K1, K2, Bb + K1, ldb, K2 >> 5, lane);
  v8f acc = a0 + a1;
  int n = lane & 15, mo = (lane >> 4) * 8;
  float bz = bias ? bias[col + n] : 0.f;
#pragma unroll
  for (int v = 0; v < 8; ++v) {
    float r = acc[v] + bz;
    if (act) r = tanhf(r);
    if (outF) outF[(size_t)(mo + v) * ldF + col + n] = r;
    if (outB) outB[(mo + v) * ldB16 + col + n] = (__bf16)r;
  }
}

// Attention energies: e[b][t] = v . tanh(enc_k[b,t,:] + q[b,:] + f[b,t,:]@Wf)
// with f = 101-tap SAME conv of previous attention weights (CCH=10).
__global__ __launch_bounds__(256) void k_energy(const float* __restrict__ aw,
                                                const float* __restrict__ encK,
                                                const float* __restrict__ q,
                                                const float* __restrict__ convk,
                                                const float* __restrict__ Wf,
                                                const float* __restrict__ vvec,
                                                const int* __restrict__ elens,
                                                float* __restrict__ e) {
  __shared__ float sk[101 * 10];
  __shared__ float swf[10 * 512];
  __shared__ float sv[512];
  for (int i = threadIdx.x; i < 101 * 10; i += 256) sk[i] = convk[i];
  for (int i = threadIdx.x; i < 10 * 512; i += 256) swf[i] = Wf[i];
  for (int i = threadIdx.x; i < 512; i += 256) sv[i] = vvec[i];
  __syncthreads();

  int gid = blockIdx.x * 256 + threadIdx.x;   // 0..8191 = b*512 + t
  int b = gid >> 9, t = gid & 511;

  float f[10];
#pragma unroll
  for (int c = 0; c < 10; ++c) f[c] = 0.f;
  const float* awb = aw + b * 512;
  for (int w = 0; w < 101; ++w) {
    int tt = t + w - 50;
    if (tt >= 0 && tt < 512) {
      float av = awb[tt];
#pragma unroll
      for (int c = 0; c < 10; ++c) f[c] += av * sk[w * 10 + c];
    }
  }

  const float* ek = encK + (size_t)gid * 512;
  const float* qb = q + b * 512;
  float acc = 0.f;
  for (int a = 0; a < 512; ++a) {
    float s = ek[a] + qb[a];
#pragma unroll
    for (int c = 0; c < 10; ++c) s += f[c] * swf[c * 512 + a];
    acc += sv[a] * tanhf(s);
  }
  if (t >= elens[b]) acc = -1e9f;
  e[gid] = acc;
}

// Softmax over T per batch + context: ctx[b][j] = sum_t aw[t] * enc_out[b][t][j]
__global__ __launch_bounds__(512) void k_softctx(const float* __restrict__ e,
                                                 const float* __restrict__ enc,
                                                 float* __restrict__ aw,
                                                 __bf16* __restrict__ ctxb) {
  int b = blockIdx.x, tid = threadIdx.x;
  __shared__ float sh[512];
  __shared__ float awsh[512];
  float ev = e[b * 512 + tid];
  sh[tid] = ev; __syncthreads();
  for (int s = 256; s > 0; s >>= 1) {
    if (tid < s) sh[tid] = fmaxf(sh[tid], sh[tid + s]);
    __syncthreads();
  }
  float m = sh[0]; __syncthreads();
  float ex = __expf(ev - m);
  sh[tid] = ex; __syncthreads();
  for (int s = 256; s > 0; s >>= 1) {
    if (tid < s) sh[tid] += sh[tid + s];
    __syncthreads();
  }
  float awv = ex / sh[0];
  aw[b * 512 + tid] = awv;
  awsh[tid] = awv; __syncthreads();

  const float* eb = enc + (size_t)b * 512 * 512;
  float acc = 0.f;
#pragma unroll 4
  for (int t = 0; t < 512; ++t) acc += awsh[t] * eb[(size_t)t * 512 + tid];
  ctxb[b * 512 + tid] = (__bf16)acc;
}

// --------------------------------- host ------------------------------------

static void launch_tr(const float* in, __bf16* out, int K, int N, hipStream_t s) {
  long total = (long)K * N;
  int blocks = (int)((total + 255) / 256);
  k_transpose_bf16<<<blocks, 256, 0, s>>>(in, out, K, N, total);
}

extern "C" void kernel_launch(void* const* d_in, const int* in_sizes, int n_in,
                              void* d_out, int out_size, void* d_ws, size_t ws_size,
                              hipStream_t stream) {
  constexpr int B = 16, L = 128, T = 512, E = 512, H = 1024, A = 512, V = 10000, BN = 1024;
  (void)in_sizes; (void)n_in; (void)out_size; (void)ws_size;

  const int*   ys    = (const int*)d_in[0];
  const int*   elens = (const int*)d_in[1];
  const float* enc   = (const float*)d_in[2];
  const float* emb   = (const float*)d_in[3];
  const float* Wx0   = (const float*)d_in[4];
  const float* Wh0   = (const float*)d_in[5];
  const float* b0    = (const float*)d_in[6];
  const float* Wx1   = (const float*)d_in[7];
  const float* Wh1   = (const float*)d_in[8];
  const float* b1    = (const float*)d_in[9];
  const float* Wk    = (const float*)d_in[10];
  const float* bk    = (const float*)d_in[11];
  const float* Wq    = (const float*)d_in[12];
  const float* Wf    = (const float*)d_in[13];
  const float* convk = (const float*)d_in[14];
  const float* vvec  = (const float*)d_in[15];
  const float* Wbn   = (const float*)d_in[16];
  const float* bbn   = (const float*)d_in[17];
  const float* Wout  = (const float*)d_in[18];
  const float* bout  = (const float*)d_in[19];
  float* out = (float*)d_out;

  char* p = (char*)d_ws;
  auto alloc = [&](size_t bytes) -> void* {
    void* r = (void*)p;
    p += (bytes + 255) & ~(size_t)255;
    return r;
  };

  __bf16* Wx0T = (__bf16*)alloc((size_t)4 * H * H * 2);           // [4096][1024]
  __bf16* Wh0T = (__bf16*)alloc((size_t)4 * H * H * 2);
  __bf16* Wx1T = (__bf16*)alloc((size_t)4 * H * H * 2);
  __bf16* Wh1T = (__bf16*)alloc((size_t)4 * H * H * 2);
  __bf16* WqT  = (__bf16*)alloc((size_t)A * H * 2);               // [512][1024]
  __bf16* WbnT = (__bf16*)alloc((size_t)BN * (H + E) * 2);        // [1024][1536]
  __bf16* WoutT= (__bf16*)alloc((size_t)V * BN * 2);              // [10000][1024]
  __bf16* WkT  = (__bf16*)alloc((size_t)A * E * 2);               // [512][512]
  __bf16* encB = (__bf16*)alloc((size_t)B * T * E * 2);
  float*  encK = (float*) alloc((size_t)B * T * A * 4);
  __bf16* yemb = (__bf16*)alloc((size_t)L * B * E * 2);           // [t][b][e]
  __bf16* h0buf[2] = {(__bf16*)alloc((size_t)B * H * 2), (__bf16*)alloc((size_t)B * H * 2)};
  __bf16* h1buf[2] = {(__bf16*)alloc((size_t)B * H * 2), (__bf16*)alloc((size_t)B * H * 2)};
  float*  c0   = (float*) alloc((size_t)B * H * 4);
  float*  c1   = (float*) alloc((size_t)B * H * 4);
  __bf16* ctxb = (__bf16*)alloc((size_t)B * E * 2);
  float*  q    = (float*) alloc((size_t)B * A * 4);
  float*  ebuf = (float*) alloc((size_t)B * T * 4);
  float*  aw   = (float*) alloc((size_t)B * T * 4);
  __bf16* ubuf = (__bf16*)alloc((size_t)B * BN * 2);

  // -------- one-time (per call) prep --------
  launch_tr(Wx0, Wx0T, H, 4 * H, stream);
  launch_tr(Wh0, Wh0T, H, 4 * H, stream);
  launch_tr(Wx1, Wx1T, H, 4 * H, stream);
  launch_tr(Wh1, Wh1T, H, 4 * H, stream);
  launch_tr(Wq,  WqT,  H, A, stream);
  launch_tr(Wbn, WbnT, H + E, BN, stream);
  launch_tr(Wout, WoutT, BN, V, stream);
  launch_tr(Wk,  WkT,  E, A, stream);

  {
    long n = (long)B * T * E;
    k_cast_bf16<<<(int)((n + 255) / 256), 256, 0, stream>>>(enc, encB, n);
  }
  k_gather_emb<<<(L * B * E) / 256, 256, 0, stream>>>(ys, emb, yemb);
  k_enck<<<dim3(B * T / 16, A / 64), 128, 0, stream>>>(encB, WkT, bk, encK);
  k_init_state<<<64, 256, 0, stream>>>(c0, c1, aw, h0buf[0], h0buf[1],
                                       h1buf[0], h1buf[1], ctxb);

  const int shLstm = 16 * (E + E + H) * 2 + 4 * 16 * 16 * 4;   // 69632 B
  const int shQ    = 16 * H * 2;                               // 32768 B
  const int shBn   = 16 * (H + E) * 2;                         // 49152 B
  const int shOut  = 16 * BN * 2;                              // 32768 B

  // -------- sequential decode: 128 steps, 7 kernels per step --------
  for (int t = 0; t < L; ++t) {
    int r = t & 1, w = r ^ 1;
    const __bf16* yt = yemb + (size_t)t * B * E;

    // LSTM layer 0: x = [y_t | ctx]  (K = 512 + 512)
    k_lstm<<<H / 16, 128, shLstm, stream>>>(yt, E, ctxb, E, h0buf[r],
                                            Wx0T, Wh0T, b0, c0, h0buf[w]);
    // LSTM layer 1: x = h0  (K = 1024)
    k_lstm<<<H / 16, 128, shLstm, stream>>>(h0buf[w], H, (const __bf16*)nullptr, 0,
                                            h1buf[r], Wx1T, Wh1T, b1, c1, h1buf[w]);
    // q = h1 @ Wq   [16,512]
    k_gemm16<<<(A / 16 + 3) / 4, 128, shQ, stream>>>(
        h1buf[w], H, (const __bf16*)nullptr, 0, WqT, (const float*)nullptr,
        q, (long)A, (__bf16*)nullptr, 0, A, 0);
    // location-aware attention energies
    k_energy<<<(B * T) / 256, 256, 0, stream>>>(aw, encK, q, convk, Wf, vvec,
                                                elens, ebuf);
    // softmax + context (fp32 -> bf16 ctx)
    k_softctx<<<B, 512, 0, stream>>>(ebuf, enc, aw, ctxb);
    // bottleneck: u = tanh([h1 | ctx] @ Wbn + bbn)  [16,1024] bf16
    k_gemm16<<<(BN / 16 + 3) / 4, 128, shBn, stream>>>(
        h1buf[w], H, ctxb, E, WbnT, bbn,
        (float*)nullptr, 0, ubuf, BN, BN, 1);
    // logits: out[:, t, :] = u @ Wout + bout   [16,10000] fp32
    k_gemm16<<<(V / 16 + 4) / 4, 128, shOut, stream>>>(
        ubuf, BN, (const __bf16*)nullptr, 0, WoutT, bout,
        out + (size_t)t * V, (long)L * V, (__bf16*)nullptr, 0, V, 0);
  }
}
